// ROIPooler_2267742732675
// MI455X (gfx1250) — matrix operand order repaired and checked by reading, not compiled, not run on hardware
//
#include <hip/hip_runtime.h>
#include <hip/hip_bf16.h>
#include <math.h>

typedef __attribute__((ext_vector_type(16))) _Float16 v16h;
typedef __attribute__((ext_vector_type(4)))  _Float16 v4h;
typedef __attribute__((ext_vector_type(2)))  _Float16 v2h;
typedef __attribute__((ext_vector_type(8)))  float    v8f;

#define CCH     256
#define NCELL   49
#define BSTRIDE 260          // 256 tap-rows + pad 4 (even => packed b32 stores stay 4B-aligned,
                             // 520B pitch keeps v4h fragment loads 8B-aligned)

// One workgroup (128 threads = 4 waves) per box.
__global__ __launch_bounds__(128)
void roi_align_wmma_kernel(const float* __restrict__ f0, const float* __restrict__ f1,
                           const float* __restrict__ f2, const float* __restrict__ f3,
                           const float* __restrict__ boxes, const int* __restrict__ bidx,
                           float* __restrict__ out)
{
    __shared__ _Float16 Alds[16 * 256];        // 8 KB   : block-diagonal weights (16 cells x 256 taps)
    __shared__ _Float16 Bt[64 * BSTRIDE];      // 32.5 KB: transposed gather panel [ch][tap-row]
    __shared__ int   rowoff[256];              // 1 KB   : per-tap-row base offset into feat
    __shared__ int   ylo_s[14], yhi_s[14], xlo_s[14], xhi_s[14];
    __shared__ float fy_s[14],  fx_s[14],  vy_s[14],  vx_s[14];

    const int box  = blockIdx.x;
    const int tid  = threadIdx.x;
    const int lane = tid & 31;
    const int wave = tid >> 5;

    // ---------- per-box setup (redundant per thread; cheap) ----------
    const float X1 = boxes[box * 4 + 0];
    const float Y1 = boxes[box * 4 + 1];
    const float X2 = boxes[box * 4 + 2];
    const float Y2 = boxes[box * 4 + 3];

    const float area = (X2 - X1) * (Y2 - Y1);
    const float sz   = sqrtf(fmaxf(area, 0.0f));
    float lv = floorf(4.0f + log2f(sz / 224.0f + 2.220446049250313e-16f));
    lv = fminf(fmaxf(lv, 2.0f), 5.0f);
    const int lvl = (int)lv - 2;                       // 0..3

    const float* feat = (lvl == 0) ? f0 : (lvl == 1) ? f1 : (lvl == 2) ? f2 : f3;
    const int   HW    = 256 >> lvl;
    const float scale = 0.25f / (float)(1 << lvl);
    feat += (size_t)bidx[box] * (size_t)HW * (size_t)HW * CCH;

    // ---------- 14 sample coords per axis ----------
    if (tid < 28) {
        const int   axis = tid / 14;                   // 0 = y, 1 = x
        const int   s    = tid % 14;
        const float lo1  = axis ? X1 : Y1;
        const float hi1  = axis ? X2 : Y2;
        const float b1   = lo1 * scale - 0.5f;
        const float b2   = hi1 * scale - 0.5f;
        const float bin  = (b2 - b1) / 7.0f;
        const float coord = b1 + ((float)s + 0.5f) * 0.5f * bin;
        const float limit = (float)HW;
        const bool  valid = (coord > -1.0f) && (coord < limit);
        const float c    = fminf(fmaxf(coord, 0.0f), limit - 1.0f);
        const float lo   = floorf(c);
        const float hi   = fminf(lo + 1.0f, limit - 1.0f);
        const float frac = c - lo;
        if (axis == 0) { ylo_s[s] = (int)lo; yhi_s[s] = (int)hi; fy_s[s] = frac; vy_s[s] = valid ? 1.0f : 0.0f; }
        else           { xlo_s[s] = (int)lo; xhi_s[s] = (int)hi; fx_s[s] = frac; vx_s[s] = valid ? 1.0f : 0.0f; }
    }
    // zero A once (diagonal block positions identical for every group)
    for (int i = tid; i < 16 * 256; i += 128) Alds[i] = (_Float16)0.0f;
    __syncthreads();

    float* outBox = out + (size_t)box * NCELL * CCH;

    const int Mr   = lane & 15;                // A row / B,D column within tile
    const int half = lane >> 4;

    // staging decomposition: thread owns channel pair (2*chp, 2*chp+1) and a 64-row band
    const int chp  = tid & 31;
    const int chA  = chp * 2;
    const int rgrp = tid >> 5;                 // 0..3 -> rows [64*rgrp, 64*rgrp+63]

#pragma unroll
    for (int g = 0; g < 4; ++g) {              // 4 groups of 16 cells cover 49 (padded)
        // ----- per group: fill diagonal weights A[m][16m+t] and tap-row offsets -----
        for (int i = tid; i < 256; i += 128) {
            const int m = i >> 4, t = i & 15;
            const int cellRaw = g * 16 + m;
            const int cell = (cellRaw < NCELL) ? cellRaw : (NCELL - 1);
            const int oy = cell / 7, ox = cell % 7;
            const int sy = (t >> 3) & 1, sx = (t >> 2) & 1, iy = (t >> 1) & 1, ix = t & 1;
            const int syi = 2 * oy + sy, sxi = 2 * ox + sx;
            const int ry = iy ? yhi_s[syi] : ylo_s[syi];
            const int rx = ix ? xhi_s[sxi] : xlo_s[sxi];
            rowoff[i] = (ry * HW + rx) * CCH;
            const float wy = (iy ? fy_s[syi] : 1.0f - fy_s[syi]) * vy_s[syi];
            const float wx = (ix ? fx_s[sxi] : 1.0f - fx_s[sxi]) * vx_s[sxi];
            float w = 0.25f * wy * wx;
            if (cellRaw >= NCELL) w = 0.0f;
            Alds[m * 256 + (m * 16 + t)] = (_Float16)w;
        }
        __syncthreads();

        // ----- hoisted A fragments for this group (chunk-invariant): 8 x v16h -----
        v16h afrag[8];
#pragma unroll
        for (int ks = 0; ks < 8; ++ks) {
#pragma unroll
            for (int e = 0; e < 8; ++e) {
                // 16-bit A (16x32) layout: lanes 0-15 -> K {0..7,16..23}; lanes 16-31 -> K {8..15,24..31}
                const int bka = ((e < 4) ? 0 : 16) + half * 8 + (e & 3) * 2;
                const int ka  = ks * 32 + bka;
                afrag[ks][2 * e]     = Alds[Mr * 256 + ka];
                afrag[ks][2 * e + 1] = Alds[Mr * 256 + ka + 1];
            }
        }

        for (int chunk = 0; chunk < 4; ++chunk) {      // 64-channel chunks
            const int chBase = chunk * 64;

            // ----- gather into transposed panel Bt[ch][tap-row] -----
            // float2 over channels (coalesced 256B/wave b64 loads), row pairs packed
            // into v2h (b32 LDS stores). 2 loads + 2 stores per 4 elements, batched x4.
            {
                const float*    src  = feat + chBase + chA;
                _Float16* const dst0 = Bt + chA * BSTRIDE;
                _Float16* const dst1 = Bt + (chA + 1) * BSTRIDE;
                for (int p0 = 0; p0 < 32; p0 += 4) {
                    float2 va[4], vb[4];
                    int    jb[4];
#pragma unroll
                    for (int q = 0; q < 4; ++q) {
                        jb[q] = rgrp * 64 + 2 * (p0 + q);
                        const int o0 = rowoff[jb[q]];
                        const int o1 = rowoff[jb[q] + 1];
                        va[q] = *(const float2*)(src + o0);
                        vb[q] = *(const float2*)(src + o1);
                    }
#pragma unroll
                    for (int q = 0; q < 4; ++q) {
                        v2h pa, pb;
                        pa[0] = (_Float16)va[q].x; pa[1] = (_Float16)vb[q].x;   // ch   : rows j, j+1
                        pb[0] = (_Float16)va[q].y; pb[1] = (_Float16)vb[q].y;   // ch+1 : rows j, j+1
                        *(v2h*)(dst0 + jb[q]) = pa;
                        *(v2h*)(dst1 + jb[q]) = pb;
                    }
                }
            }
            __syncthreads();

            // ----- 8 x V_WMMA_F32_16X16X32_F16, one 16-ch N-tile per wave -----
            v8f acc = {};
            const int nl = wave * 16 + Mr;             // channel within chunk
#pragma unroll
            for (int ks = 0; ks < 8; ++ks) {
                // B (32x16) layout: lanes 0-15 -> K {0..7,16..23}, lanes 16-31 -> K {8..15,24..31} (matches A).
                // With Bt channel-major, a lane's 16 B values are two runs of 8 consecutive f16.
                const v4h* bp = (const v4h*)(Bt + nl * BSTRIDE + ks * 32 + half * 8);
                v16h b;
#pragma unroll
                for (int q = 0; q < 4; ++q) {
                    const v4h tq = bp[q + ((q >= 2) ? 2 : 0)];
                    b[4 * q + 0] = tq[0]; b[4 * q + 1] = tq[1];
                    b[4 * q + 2] = tq[2]; b[4 * q + 3] = tq[3];
                }
                acc = __builtin_amdgcn_wmma_f32_16x16x32_f16(
                    /*neg_a=*/false, afrag[ks], /*neg_b=*/false, b,
                    /*c_mod=*/(short)0, acc, /*reuse_a=*/false, /*reuse_b=*/false);
            }

            // ----- store D (16x16 f32 layout: lane -> N, VGPR v -> M = 8*half + v) -----
            // g is compile-time (outer loop unrolled): groups 0..2 store unconditionally.
            const int mtop = half * 8;
#pragma unroll
            for (int v = 0; v < 8; ++v) {
                const int cell = g * 16 + mtop + v;
                if (g < 3 || cell < NCELL)
                    outBox[(size_t)cell * CCH + chBase + wave * 16 + Mr] = acc[v];
            }
            __syncthreads();
        }
    }
}

extern "C" void kernel_launch(void* const* d_in, const int* in_sizes, int n_in,
                              void* d_out, int out_size, void* d_ws, size_t ws_size,
                              hipStream_t stream) {
    const float* f0    = (const float*)d_in[0];   // feat2: (2,256,256,256)
    const float* f1    = (const float*)d_in[1];   // feat3: (2,128,128,256)
    const float* f2    = (const float*)d_in[2];   // feat4: (2, 64, 64,256)
    const float* f3    = (const float*)d_in[3];   // feat5: (2, 32, 32,256)
    const float* boxes = (const float*)d_in[4];   // (512,4)
    const int*   bidx  = (const int*)d_in[5];     // (512,)
    float*       out   = (float*)d_out;           // (512,7,7,256)

    const int M = in_sizes[4] / 4;                // 512 boxes
    roi_align_wmma_kernel<<<M, 128, 0, stream>>>(f0, f1, f2, f3, boxes, bidx, out);
}
